// LightAttention_22969485099845
// MI455X (gfx1250) — matrix operand order repaired and checked by reading, not compiled, not run on hardware
//
#include <hip/hip_runtime.h>

typedef __bf16 bf16;
typedef __attribute__((ext_vector_type(16))) __bf16 v16bf;
typedef __attribute__((ext_vector_type(8)))  float  v8f;
typedef __attribute__((ext_vector_type(4)))  unsigned int v4u;

union Frag   { v16bf v; v4u q[2]; };
union Pack16 { bf16 h[16]; v4u q[2]; };
union Pair   { bf16 h[2]; unsigned int u; };

// Load 8 consecutive elements as float, using full-width vector loads.
template<typename T>
__device__ __forceinline__ void load8f(const T* __restrict__ p, float out[8]) {
    if constexpr (sizeof(T) == 4) {
        const float4 a = *(const float4*)(p);
        const float4 b = *(const float4*)(p + 4);
        out[0] = a.x; out[1] = a.y; out[2] = a.z; out[3] = a.w;
        out[4] = b.x; out[5] = b.y; out[6] = b.z; out[7] = b.w;
    } else {
        union { v4u q; bf16 h[8]; } t;
        t.q = *(const v4u*)(p);
        #pragma unroll
        for (int i = 0; i < 8; ++i) out[i] = (float)t.h[i];
    }
}

// ---------------------------------------------------------------------------
// bf16-WMMA GEMM with LDS double buffering:
//   C[M,N] = A[M,K] * B[K,N] (+ bias[N] if HAS_BIAS)
//   TRANSA: A stored [K][M] in memory (for Ks^T @ V)
//   B stored [K][N]; staged K-transposed into LDS as Bs[n][k]
// Block tile 128x128, BK=32, 256 threads = 8 waves, wave tile 32x64.
// ---------------------------------------------------------------------------
template<typename TA, typename TB, typename TOUT, bool TRANSA, bool HAS_BIAS>
__global__ __launch_bounds__(256)
void gemm_wmma_kernel(const TA* __restrict__ Ag, const TB* __restrict__ Bg,
                      const float* __restrict__ bias, TOUT* __restrict__ Cg,
                      int M, int N, int K,
                      long long sA, long long sB, long long sC)
{
    constexpr int BM = 128, BN = 128, BK = 32, LDSS = 40; // 80B rows, 16B aligned
    __shared__ bf16 As[2][BM * LDSS];
    __shared__ bf16 Bs[2][BN * LDSS];

    const int tid  = threadIdx.x;
    const int lane = tid & 31;
    const int wid  = tid >> 5;
    const int wm   = wid & 3;   // 4 waves along M -> 32 rows each
    const int wn   = wid >> 2;  // 2 waves along N -> 64 cols each

    const int m0 = blockIdx.y * BM;
    const int n0 = blockIdx.x * BN;
    const long long z = blockIdx.z;
    Ag += z * sA;
    Bg += z * sB;
    Cg += z * sC;

    const int lda = TRANSA ? M : K;
    const int ldb = N;

    // staging decompositions
    const int rowA  = tid >> 1;         // non-trans A: 0..127
    const int kOffA = (tid & 1) * 16;   // non-trans A: 0 or 16
    const int kl2   = (tid >> 4) * 2;   // transposed paths: even k, 0..30
    const int xOff  = (tid & 15) * 8;   // transposed paths: 8-wide m/n segment

    auto stageA = [&](int k0, int buf) {
        if constexpr (!TRANSA) {
            const TA* src = Ag + (long long)(m0 + rowA) * lda + (k0 + kOffA);
            float f[8];
            Pack16 pk;
            load8f(src, f);
            #pragma unroll
            for (int i = 0; i < 8; ++i) pk.h[i] = (bf16)f[i];
            load8f(src + 8, f);
            #pragma unroll
            for (int i = 0; i < 8; ++i) pk.h[8 + i] = (bf16)f[i];
            bf16* dst = &As[buf][rowA * LDSS + kOffA];
            *(v4u*)(dst)     = pk.q[0];
            *(v4u*)(dst + 8) = pk.q[1];
        } else {
            float f0[8], f1[8];
            load8f(Ag + (long long)(k0 + kl2)     * lda + (m0 + xOff), f0);
            load8f(Ag + (long long)(k0 + kl2 + 1) * lda + (m0 + xOff), f1);
            #pragma unroll
            for (int e = 0; e < 8; ++e) {
                Pair p; p.h[0] = (bf16)f0[e]; p.h[1] = (bf16)f1[e];
                *(unsigned int*)&As[buf][(xOff + e) * LDSS + kl2] = p.u;
            }
        }
    };
    auto stageB = [&](int k0, int buf) {
        float f0[8], f1[8];
        load8f(Bg + (long long)(k0 + kl2)     * ldb + (n0 + xOff), f0);
        load8f(Bg + (long long)(k0 + kl2 + 1) * ldb + (n0 + xOff), f1);
        #pragma unroll
        for (int e = 0; e < 8; ++e) {
            Pair p; p.h[0] = (bf16)f0[e]; p.h[1] = (bf16)f1[e];
            *(unsigned int*)&Bs[buf][(xOff + e) * LDSS + kl2] = p.u;
        }
    };

    v8f acc[2][4];
    #pragma unroll
    for (int mt = 0; mt < 2; ++mt)
        #pragma unroll
        for (int nt = 0; nt < 4; ++nt)
            acc[mt][nt] = (v8f)0.0f;

    // fragment layout per ISA:
    // A 16x32: lanes 0-15 row=lane, K {0..7,16..23}; lanes 16-31 K {8..15,24..31}
    // B 32x16: lanes 0-15 col=lane, K {0..7,16..23}; lanes 16-31 K {8..15,24..31}
    const int hr = lane & 15;
    const int hk = (lane >> 4) * 8;
    const int kb = (lane >> 4) * 16;

    auto compute = [&](int buf) {
        Frag a[2], b[4];
        #pragma unroll
        for (int mt = 0; mt < 2; ++mt) {
            const int row = wm * 32 + mt * 16 + hr;
            a[mt].q[0] = *(const v4u*)&As[buf][row * LDSS + hk];
            a[mt].q[1] = *(const v4u*)&As[buf][row * LDSS + 16 + hk];
        }
        #pragma unroll
        for (int nt = 0; nt < 4; ++nt) {
            const int col = wn * 64 + nt * 16 + hr;
            b[nt].q[0] = *(const v4u*)&Bs[buf][col * LDSS + hk];
            b[nt].q[1] = *(const v4u*)&Bs[buf][col * LDSS + 16 + hk];
        }
        #pragma unroll
        for (int mt = 0; mt < 2; ++mt)
            #pragma unroll
            for (int nt = 0; nt < 4; ++nt)
                acc[mt][nt] = __builtin_amdgcn_wmma_f32_16x16x32_bf16(
                    false, a[mt].v, false, b[nt].v,
                    (short)0, acc[mt][nt], false, false);
    };

    // double-buffered main loop: one barrier per BK step
    const int nk = K / BK;
    stageA(0, 0);
    stageB(0, 0);
    __syncthreads();
    for (int kt = 0; kt < nk; ++kt) {
        const int cur = kt & 1;
        if (kt + 1 < nk) {
            stageA((kt + 1) * BK, cur ^ 1);
            stageB((kt + 1) * BK, cur ^ 1);
        }
        compute(cur);
        __syncthreads();
    }

    // epilogue: C/D layout: VGPR i -> row i + 8*(lane>>4), col lane&15
    const int lr = (lane >> 4) * 8;
    const int lc = lane & 15;
    #pragma unroll
    for (int mt = 0; mt < 2; ++mt) {
        #pragma unroll
        for (int nt = 0; nt < 4; ++nt) {
            const int grb = m0 + wm * 32 + mt * 16 + lr;
            const int gc  = n0 + wn * 64 + nt * 16 + lc;
            float bv = 0.0f;
            if constexpr (HAS_BIAS) bv = bias[gc];
            #pragma unroll
            for (int i = 0; i < 8; ++i) {
                const float v = acc[mt][nt][i] + bv;
                Cg[(long long)(grb + i) * N + gc] = (TOUT)v;
            }
        }
    }
}

// ---------------------------------------------------------------------------
// Column softmax over sequence axis of Y[B][N][D], two-pass chunked online.
// ---------------------------------------------------------------------------
__global__ __launch_bounds__(512)
void softmax_stats_kernel(const bf16* __restrict__ Y, float* __restrict__ stats,
                          float scale, int Nrows, int D, int rowsPerChunk, int NCH)
{
    const int b  = blockIdx.x;
    const int ch = blockIdx.y;
    const int d  = threadIdx.x;
    const bf16* p = Y + ((long long)b * Nrows + (long long)ch * rowsPerChunk) * D + d;
    float m = -__builtin_inff();
    float s = 0.0f;
    for (int r = 0; r < rowsPerChunk; ++r) {
        const float x = (float)p[(long long)r * D] * scale;
        if (x > m) { s = s * __expf(m - x) + 1.0f; m = x; }
        else       { s += __expf(x - m); }
    }
    float* st = stats + ((long long)(b * NCH + ch) * D + d) * 2;
    st[0] = m;
    st[1] = s;
}

__global__ __launch_bounds__(512)
void softmax_norm_kernel(bf16* __restrict__ Y, const float* __restrict__ stats,
                         float scale, int Nrows, int D, int rowsPerChunk, int NCH)
{
    const int b  = blockIdx.x;
    const int ch = blockIdx.y;
    const int d  = threadIdx.x;
    float m = -__builtin_inff();
    float s = 0.0f;
    for (int c = 0; c < NCH; ++c) {
        const float* st = stats + ((long long)(b * NCH + c) * D + d) * 2;
        const float mc = st[0], sc = st[1];
        const float mn = m > mc ? m : mc;
        s = s * __expf(m - mn) + sc * __expf(mc - mn);
        m = mn;
    }
    const float inv = 1.0f / s;
    bf16* p = Y + ((long long)b * Nrows + (long long)ch * rowsPerChunk) * D + d;
    for (int r = 0; r < rowsPerChunk; ++r) {
        const float x = (float)p[(long long)r * D] * scale;
        p[(long long)r * D] = (bf16)(__expf(x - m) * inv);
    }
}

// ---------------------------------------------------------------------------
extern "C" void kernel_launch(void* const* d_in, const int* in_sizes, int n_in,
                              void* d_out, int out_size, void* d_ws, size_t ws_size,
                              hipStream_t stream)
{
    (void)in_sizes; (void)n_in; (void)out_size; (void)ws_size;

    const float* x_q = (const float*)d_in[0];
    const float* x_k = (const float*)d_in[1];
    const float* x_v = (const float*)d_in[2];
    const float* W_q = (const float*)d_in[3];
    const float* b_q = (const float*)d_in[4];
    const float* W_k = (const float*)d_in[5];
    const float* b_k = (const float*)d_in[6];
    const float* W_v = (const float*)d_in[7];
    const float* b_v = (const float*)d_in[8];
    float* Z = (float*)d_out;

    constexpr int B = 8, N = 4096, D = 512;
    constexpr int NCH = 32, RPC = N / NCH;
    const float scale = 0.21022410381342863f; // 512^(-1/4)

    // workspace layout
    char* ws = (char*)d_ws;
    const size_t SZ   = (size_t)B * N * D * sizeof(bf16);   // 32 MiB
    const size_t BMSZ = (size_t)B * D * D * sizeof(bf16);   //  4 MiB
    bf16* Qbf = (bf16*)(ws);
    bf16* Kbf = (bf16*)(ws + SZ);
    bf16* Vbf = (bf16*)(ws + 2 * SZ);
    bf16* Bm  = (bf16*)(ws + 3 * SZ);
    float* stQ = (float*)(ws + 3 * SZ + BMSZ);
    float* stK = stQ + (size_t)B * NCH * D * 2;

    const long long ND = (long long)N * D;
    const long long DD = (long long)D * D;

    // 1) projections: [B*N,512] @ [512,512] + bias -> bf16
    {
        dim3 g(D / 128, (B * N) / 128, 1), blk(256);
        gemm_wmma_kernel<float, float, bf16, false, true><<<g, blk, 0, stream>>>(
            x_q, W_q, b_q, Qbf, B * N, D, D, 0, 0, 0);
        gemm_wmma_kernel<float, float, bf16, false, true><<<g, blk, 0, stream>>>(
            x_k, W_k, b_k, Kbf, B * N, D, D, 0, 0, 0);
        gemm_wmma_kernel<float, float, bf16, false, true><<<g, blk, 0, stream>>>(
            x_v, W_v, b_v, Vbf, B * N, D, D, 0, 0, 0);
    }

    // 2) column softmax over sequence axis on Q and K (in place, bf16)
    {
        dim3 g(B, NCH), blk(D);
        softmax_stats_kernel<<<g, blk, 0, stream>>>(Qbf, stQ, scale, N, D, RPC, NCH);
        softmax_stats_kernel<<<g, blk, 0, stream>>>(Kbf, stK, scale, N, D, RPC, NCH);
        softmax_norm_kernel<<<g, blk, 0, stream>>>(Qbf, stQ, scale, N, D, RPC, NCH);
        softmax_norm_kernel<<<g, blk, 0, stream>>>(Kbf, stK, scale, N, D, RPC, NCH);
    }

    // 3) Bm[b] = Ks^T @ V : M=512, N=512, K=4096 (A transposed from [n][d])
    {
        dim3 g(D / 128, D / 128, B), blk(256);
        gemm_wmma_kernel<bf16, bf16, bf16, true, false><<<g, blk, 0, stream>>>(
            Kbf, Vbf, nullptr, Bm, D, D, N, ND, ND, DD);
    }

    // 4) Z[b] = A @ Bm : M=4096, N=512, K=512 -> fp32 output
    {
        dim3 g(D / 128, N / 128, B), blk(256);
        gemm_wmma_kernel<bf16, bf16, float, false, false><<<g, blk, 0, stream>>>(
            Qbf, Bm, nullptr, Z, N, D, D, ND, DD, ND);
    }
}
